// DGATEncoder_20572893347936
// MI455X (gfx1250) — matrix operand order repaired and checked by reading, not compiled, not run on hardware
//
#include <hip/hip_runtime.h>
#include <cstdint>
#include <cstddef>

// ---------------------------------------------------------------------------
// DGAT encoder for MI455X (gfx1250, wave32, WMMA).
//  - All GEMMs run on v_wmma_f32_16x16x32_bf16 (f32 accumulate).
//  - gather∘matmul commuted: project all messages once, gather projected rows.
//  - GAT score separability: score(i,m,h) = sq[i,h] + sk[bgraph[i,m],h] + abias[h].
//  - GEMM K-loop split into branch-free vector main loop + guarded tail.
// ---------------------------------------------------------------------------

#define HID    256
#define HEADS  8
#define MAXNN  6
#define NEGV   (-1e18f)

typedef __bf16 bf16_t;
typedef __attribute__((ext_vector_type(16))) __bf16 v16bf;
typedef __attribute__((ext_vector_type(8)))  float  v8f;

__device__ __forceinline__ bf16_t tobf(float f) { return (bf16_t)f; }
__device__ __forceinline__ float sigm(float x) { return 1.0f / (1.0f + __expf(-x)); }

// ---- transpose weight [Ksrc x N] (rows rowOff..rowOff+Ksrc-1) -> bf16 WT[N x ldp],
//      zero-padded in K so GEMM B loads never need guards. -----------------------
__global__ void transpose_w(const float* __restrict__ W, int rowOff, int Ksrc, int N,
                            int ldp, bf16_t* __restrict__ WT) {
  int idx = blockIdx.x * blockDim.x + threadIdx.x;
  int total = N * ldp;
  if (idx >= total) return;
  int n = idx / ldp, k = idx - n * ldp;
  float v = (k < Ksrc) ? W[(size_t)(rowOff + k) * N + n] : 0.0f;
  WT[idx] = tobf(v);
}

// ---- generic GEMM: C[M x N] = A[M x K](f32,row-major,lda) @ W + bias
//      W given pre-transposed bf16: WT[n*ldp + k], ldp = roundup32(K), zero padded.
//      One wave owns a 16x64 output tile (4 WMMA accumulators).
//      Out-of-range rows are clamped (their D rows are never stored), so the
//      main loop is completely branch-free with full EXEC. ------------------------
__global__ void __launch_bounds__(256)
gemm_wmma(const float* __restrict__ A, int lda,
          const bf16_t* __restrict__ WT, int ldp,
          const float* __restrict__ bias,
          float* __restrict__ C, int M, int N, int K) {
  int wave = (blockIdx.x * blockDim.x + threadIdx.x) >> 5;
  int lane = threadIdx.x & 31;
  int ngroups = N >> 6;               // 64-wide col groups
  int mtile = wave / ngroups;
  int ng    = wave - mtile * ngroups;
  int m0 = mtile * 16;
  if (m0 >= M) return;
  int n0 = ng * 64;

  int row = m0 + (lane & 15);
  if (row >= M) row = M - 1;          // clamp: those D rows are never stored
  const float* Arow = A + (size_t)row * lda;
  int khA = (lane >> 4) * 16;         // A: e   -> kt+khA+e
  int khB = (lane >> 4) * 16;         // B: e   -> kt+khB+e
  int cl  = lane & 15;
  const bf16_t* Bbase = WT + (size_t)(n0 + cl) * ldp + khB;

  v8f acc[4];
  acc[0] = {}; acc[1] = {}; acc[2] = {}; acc[3] = {};

  const int kvec = ((lda & 3) == 0) ? (K & ~31) : 0;   // vectorizable prefix
  int kt = 0;
  for (; kt < kvec; kt += 32) {       // branch-free hot loop
    v16bf a;
    {
      const float4* p0 = (const float4*)(Arow + kt + khA);
      const float4* p1 = (const float4*)(Arow + kt + khA + 8);
      float4 x0 = p0[0], x1 = p0[1], y0 = p1[0], y1 = p1[1];
      a[0]  = tobf(x0.x); a[1]  = tobf(x0.y); a[2]  = tobf(x0.z); a[3]  = tobf(x0.w);
      a[4]  = tobf(x1.x); a[5]  = tobf(x1.y); a[6]  = tobf(x1.z); a[7]  = tobf(x1.w);
      a[8]  = tobf(y0.x); a[9]  = tobf(y0.y); a[10] = tobf(y0.z); a[11] = tobf(y0.w);
      a[12] = tobf(y1.x); a[13] = tobf(y1.y); a[14] = tobf(y1.z); a[15] = tobf(y1.w);
    }
#pragma unroll
    for (int s = 0; s < 4; ++s) {
      v16bf b = *(const v16bf*)(Bbase + (size_t)s * 16 * ldp + kt);
      acc[s] = __builtin_amdgcn_wmma_f32_16x16x32_bf16(
          false, a, false, b, (short)0, acc[s], false, false);
    }
  }
  for (; kt < K; kt += 32) {          // guarded tail (K remainder / odd lda)
    v16bf a;
#pragma unroll
    for (int e = 0; e < 16; ++e) {
      int k = kt + khA + e;
      float v = (k < K) ? Arow[k] : 0.0f;
      a[e] = tobf(v);
    }
#pragma unroll
    for (int s = 0; s < 4; ++s) {
      v16bf b = *(const v16bf*)(Bbase + (size_t)s * 16 * ldp + kt);
      acc[s] = __builtin_amdgcn_wmma_f32_16x16x32_bf16(
          false, a, false, b, (short)0, acc[s], false, false);
    }
  }

  // ---- epilogue: D layout -> M = e + 8*(lane>=16), N = lane&15 ----
  int crow = m0 + 8 * (lane >> 4);
#pragma unroll
  for (int s = 0; s < 4; ++s) {
    int col = n0 + s * 16 + cl;
    float bv = bias ? bias[col] : 0.0f;
#pragma unroll
    for (int e = 0; e < 8; ++e) {
      int r = crow + e;
      if (r < M) C[(size_t)r * N + col] = acc[s][e] + bv;
    }
  }
}

// ---- per-row attention scalar: out[row,head] = sum_t leaky(X[row,head*32+t]) *
//      alpha[head*64 + aoff + t]   (aoff=0 for q-half, 32 for k-half) -------------
__global__ void __launch_bounds__(256)
attn_scalar(const float* __restrict__ X, const float* __restrict__ alpha, int aoff,
            float* __restrict__ out, int rows) {
  int wave = (blockIdx.x * blockDim.x + threadIdx.x) >> 5;
  int lane = threadIdx.x & 31;
  if (wave >= rows) return;
  int head = lane >> 2;
  int t0 = (lane & 3) * 8;
  const float* xp = X + (size_t)wave * HID + head * 32 + t0;
  const float* ap = alpha + head * 64 + aoff + t0;
  float p = 0.0f;
#pragma unroll
  for (int e = 0; e < 8; ++e) {
    float x = xp[e];
    x = (x >= 0.0f) ? x : 0.01f * x;
    p += x * ap[e];
  }
  p += __shfl_xor(p, 1);
  p += __shfl_xor(p, 2);
  if ((lane & 3) == 0) out[(size_t)wave * HEADS + head] = p;
}

// ---- row mask: flags[row] = (sum over HID of X[row,:] == 0) -----------------------
__global__ void __launch_bounds__(256)
row_mask(const float* __restrict__ X, int* __restrict__ flags, int rows) {
  int wave = (blockIdx.x * blockDim.x + threadIdx.x) >> 5;
  int lane = threadIdx.x & 31;
  if (wave >= rows) return;
  const float* p = X + (size_t)wave * HID + lane * 8;
  float s = 0.0f;
#pragma unroll
  for (int e = 0; e < 8; ++e) s += p[e];
  s += __shfl_xor(s, 1);  s += __shfl_xor(s, 2);  s += __shfl_xor(s, 4);
  s += __shfl_xor(s, 8);  s += __shfl_xor(s, 16);
  if (lane == 0) flags[wave] = (s == 0.0f) ? 1 : 0;
}

// ---- gather + softmax over 6 neighbors + weighted V sum. One wave per target row. --
__global__ void __launch_bounds__(256)
attn_gather(const int* __restrict__ graph, const float* __restrict__ sq,
            const float* __restrict__ sk, const float* __restrict__ abias,
            const int* __restrict__ flags, const float* __restrict__ V,
            float* __restrict__ S, int rows) {
  int wave = (blockIdx.x * blockDim.x + threadIdx.x) >> 5;
  int lane = threadIdx.x & 31;
  if (wave >= rows) return;
  int head = lane >> 2;
  int d0 = lane * 8;
  float sqv = sq[(size_t)wave * HEADS + head];
  float ab = abias[head];
  int jm[MAXNN]; float sc[MAXNN];
  float mx = -3.0e38f;
#pragma unroll
  for (int m = 0; m < MAXNN; ++m) {
    int j = graph[(size_t)wave * MAXNN + m];
    jm[m] = j;
    float s = sqv + sk[(size_t)j * HEADS + head] + ab;
    if (flags[j]) s = NEGV;
    sc[m] = s;
    mx = fmaxf(mx, s);
  }
  float sum = 0.0f;
#pragma unroll
  for (int m = 0; m < MAXNN; ++m) { sc[m] = __expf(sc[m] - mx); sum += sc[m]; }
  float inv = 1.0f / sum;
  float4 a0 = {0, 0, 0, 0}, a1 = {0, 0, 0, 0};
#pragma unroll
  for (int m = 0; m < MAXNN; ++m) {
    const float4* vp = (const float4*)(V + (size_t)jm[m] * HID + d0);
    float4 v0 = vp[0], v1 = vp[1];
    float w = sc[m] * inv;
    a0.x = fmaf(w, v0.x, a0.x); a0.y = fmaf(w, v0.y, a0.y);
    a0.z = fmaf(w, v0.z, a0.z); a0.w = fmaf(w, v0.w, a0.w);
    a1.x = fmaf(w, v1.x, a1.x); a1.y = fmaf(w, v1.y, a1.y);
    a1.z = fmaf(w, v1.z, a1.z); a1.w = fmaf(w, v1.w, a1.w);
  }
  float4* sp = (float4*)(S + (size_t)wave * HID + d0);
  sp[0] = a0; sp[1] = a1;
}

// ---- elementwise kernels -----------------------------------------------------------
__global__ void zero_f(float* p, int n) {
  int i = blockIdx.x * blockDim.x + threadIdx.x;
  if (i < n) p[i] = 0.0f;
}
__global__ void copy_f(const float* __restrict__ s, float* __restrict__ d, int n) {
  int i = blockIdx.x * blockDim.x + threadIdx.x;
  if (i < n) d[i] = s[i];
}
// z = sigmoid(Fz+T1) -> T1 ; r = sigmoid(Fr+T2) ; T2 = r*S
__global__ void gru1(const float* __restrict__ Fz, float* __restrict__ T1,
                     const float* __restrict__ Fr, float* __restrict__ T2,
                     const float* __restrict__ S, int n) {
  int i = blockIdx.x * blockDim.x + threadIdx.x;
  if (i >= n) return;
  float z = sigm(Fz[i] + T1[i]);
  float r = sigm(Fr[i] + T2[i]);
  T1[i] = z;
  T2[i] = r * S[i];
}
// h = (1-z)*S + z*tanh(Fh+T3) ; h[row 0] = 0
__global__ void gru2(const float* __restrict__ Fh, const float* __restrict__ T3,
                     const float* __restrict__ z, const float* __restrict__ S,
                     float* __restrict__ h, int n) {
  int i = blockIdx.x * blockDim.x + threadIdx.x;
  if (i >= n) return;
  float ph = tanhf(Fh[i] + T3[i]);
  float zz = z[i];
  float hv = (1.0f - zz) * S[i] + zz * ph;
  if (i < HID) hv = 0.0f;   // row 0 masked
  h[i] = hv;
}
// out = gelu(Fo+G) * mask[row]   (exact erf gelu)
__global__ void out_gelu(const float* __restrict__ Fo, const float* __restrict__ G,
                         const float* __restrict__ mask, float* __restrict__ out, int n) {
  int i = blockIdx.x * blockDim.x + threadIdx.x;
  if (i >= n) return;
  float x = Fo[i] + G[i];
  float g = 0.5f * x * (1.0f + erff(x * 0.70710678118654752f));
  out[i] = g * mask[i >> 8];
}

// ---------------------------------------------------------------------------
extern "C" void kernel_launch(void* const* d_in, const int* in_sizes, int n_in,
                              void* d_out, int out_size, void* d_ws, size_t ws_size,
                              hipStream_t stream) {
  const float* fnode = (const float*)d_in[0];
  const float* fmess = (const float*)d_in[1];
  const int*   agraph = (const int*)d_in[2];
  const int*   bgraph = (const int*)d_in[3];
  const float* maskp  = (const float*)d_in[4];
  const float* Wq_w = (const float*)d_in[5];  const float* Wq_b = (const float*)d_in[6];
  const float* Wk_w = (const float*)d_in[7];  const float* Wk_b = (const float*)d_in[8];
  const float* Wv_w = (const float*)d_in[9];  const float* Wv_b = (const float*)d_in[10];
  const float* alpha = (const float*)d_in[11]; const float* abias = (const float*)d_in[12];
  const float* Wz_w = (const float*)d_in[13]; const float* Wz_b = (const float*)d_in[14];
  const float* Wr_w = (const float*)d_in[15];
  const float* Ur_w = (const float*)d_in[16]; const float* Ur_b = (const float*)d_in[17];
  const float* Wh_w = (const float*)d_in[18]; const float* Wh_b = (const float*)d_in[19];
  const float* eWq_w = (const float*)d_in[20]; const float* eWq_b = (const float*)d_in[21];
  const float* eWk_w = (const float*)d_in[22]; const float* eWk_b = (const float*)d_in[23];
  const float* eWv_w = (const float*)d_in[24]; const float* eWv_b = (const float*)d_in[25];
  const float* ealpha = (const float*)d_in[26]; const float* eabias = (const float*)d_in[27];
  const float* Wo_w = (const float*)d_in[28]; const float* Wo_b = (const float*)d_in[29];

  const int NM  = in_sizes[3] / MAXNN;   // 30000 messages
  const int NN  = in_sizes[2] / MAXNN;   // 15000 nodes
  const int INF = in_sizes[1] / NM;      // 112
  const int NF  = in_sizes[0] / NN;      // 98
  const int KzB = (in_sizes[13] / HID) - INF;  // 256
  const int KoB = (in_sizes[28] / HID) - NF;   // 112
  const int INFp = (INF + 31) & ~31;     // 128
  const int NFp  = (NF + 31) & ~31;      // 128
  const int KoBp = (KoB + 31) & ~31;     // 128

  // ---- bump allocator over d_ws ----
  size_t off = 0;
  auto alloc = [&](size_t bytes) -> char* {
    char* p = (char*)d_ws + off;
    off += (bytes + 255) & ~(size_t)255;
    return p;
  };
  // transposed bf16 weights (padded in K)
  bf16_t* WqT    = (bf16_t*)alloc((size_t)HID * INFp * 2);
  bf16_t* WkT    = (bf16_t*)alloc((size_t)HID * HID * 2);
  bf16_t* WvT    = (bf16_t*)alloc((size_t)HID * HID * 2);
  bf16_t* WzTopT = (bf16_t*)alloc((size_t)HID * INFp * 2);
  bf16_t* WzBotT = (bf16_t*)alloc((size_t)HID * KzB * 2);
  bf16_t* WrT    = (bf16_t*)alloc((size_t)HID * INFp * 2);
  bf16_t* UrT    = (bf16_t*)alloc((size_t)HID * HID * 2);
  bf16_t* WhTopT = (bf16_t*)alloc((size_t)HID * INFp * 2);
  bf16_t* WhBotT = (bf16_t*)alloc((size_t)HID * KzB * 2);
  bf16_t* eWqT   = (bf16_t*)alloc((size_t)HID * NFp * 2);
  bf16_t* eWkT   = (bf16_t*)alloc((size_t)HID * HID * 2);
  bf16_t* eWvT   = (bf16_t*)alloc((size_t)HID * HID * 2);
  bf16_t* WoTopT = (bf16_t*)alloc((size_t)HID * NFp * 2);
  bf16_t* WoBotT = (bf16_t*)alloc((size_t)HID * KoBp * 2);
  // big f32 buffers
  const size_t BM = (size_t)NM * HID * 4;
  float* Fz   = (float*)alloc(BM);
  float* Fr   = (float*)alloc(BM);
  float* Fh   = (float*)alloc(BM);
  float* hbuf = (float*)alloc(BM);
  float* bufA = (float*)alloc(BM);
  float* bufB = (float*)alloc(BM);
  float* Sbuf = (float*)alloc(BM);
  float* bufD = (float*)alloc(BM);
  float* sq   = (float*)alloc((size_t)NM * HEADS * 4);
  float* sk   = (float*)alloc((size_t)NM * HEADS * 4);
  float* sqn  = (float*)alloc((size_t)NN * HEADS * 4);
  int*   flags = (int*)alloc((size_t)NM * 4);

  auto trs = [&](const float* W, int rowOff, int Ksrc, int ldp, bf16_t* WT) {
    int total = HID * ldp;
    transpose_w<<<(total + 255) / 256, 256, 0, stream>>>(W, rowOff, Ksrc, HID, ldp, WT);
  };
  auto gemm = [&](const float* A, int lda, const bf16_t* WT, int ldp,
                  const float* bias, float* C, int M, int K) {
    int waves = ((M + 15) / 16) * (HID / 64);
    gemm_wmma<<<(waves + 7) / 8, 256, 0, stream>>>(A, lda, WT, ldp, bias, C, M, HID, K);
  };
  auto rowsGrid = [](int rows) { return dim3((rows + 7) / 8); };

  // ---- phase 0: weight transpose + bf16 conversion (once) ----
  trs(Wq_w, 0, INF, INFp, WqT);
  trs(Wk_w, 0, HID, HID, WkT);
  trs(Wv_w, 0, HID, HID, WvT);
  trs(Wz_w, 0, INF, INFp, WzTopT);
  trs(Wz_w, INF, KzB, KzB, WzBotT);
  trs(Wr_w, 0, INF, INFp, WrT);
  trs(Ur_w, 0, HID, HID, UrT);
  trs(Wh_w, 0, INF, INFp, WhTopT);
  trs(Wh_w, INF, KzB, KzB, WhBotT);
  trs(eWq_w, 0, NF, NFp, eWqT);
  trs(eWk_w, 0, HID, HID, eWkT);
  trs(eWv_w, 0, HID, HID, eWvT);
  trs(Wo_w, 0, NF, NFp, WoTopT);
  trs(Wo_w, NF, KoB, KoBp, WoBotT);

  // ---- phase 1: depth-invariant precompute ----
  int nElemM = NM * HID;
  zero_f<<<(nElemM + 255) / 256, 256, 0, stream>>>(hbuf, nElemM);
  gemm(fmess, INF, WqT, INFp, Wq_b, bufA, NM, INF);              // Q
  attn_scalar<<<rowsGrid(NM), 256, 0, stream>>>(bufA, alpha, 0, sq, NM);
  gemm(fmess, INF, WzTopT, INFp, nullptr, Fz, NM, INF);          // fmess @ Wz_top
  gemm(fmess, INF, WrT, INFp, nullptr, Fr, NM, INF);             // fmess @ Wr
  gemm(fmess, INF, WhTopT, INFp, nullptr, Fh, NM, INF);          // fmess @ Wh_top
  gemm(fnode, NF, eWqT, NFp, eWq_b, bufA, NN, NF);               // enc Q
  attn_scalar<<<rowsGrid(NN), 256, 0, stream>>>(bufA, ealpha, 0, sqn, NN);

  // ---- phase 2: DEPTH iterations of DGAT-GRU ----
  for (int depth = 0; depth < 4; ++depth) {
    gemm(hbuf, HID, WkT, HID, Wk_b, bufA, NM, HID);              // K = h@Wk+b
    attn_scalar<<<rowsGrid(NM), 256, 0, stream>>>(bufA, alpha, 32, sk, NM);
    row_mask<<<rowsGrid(NM), 256, 0, stream>>>(hbuf, flags, NM);
    gemm(hbuf, HID, WvT, HID, Wv_b, bufB, NM, HID);              // V = h@Wv+b
    attn_gather<<<rowsGrid(NM), 256, 0, stream>>>(bgraph, sq, sk, abias, flags,
                                                  bufB, Sbuf, NM);
    gemm(Sbuf, HID, WzBotT, KzB, Wz_b, bufA, NM, KzB);           // S@Wz_bot + bz
    gemm(Sbuf, HID, UrT, HID, Ur_b, bufD, NM, HID);              // S@Ur + br
    gru1<<<(nElemM + 255) / 256, 256, 0, stream>>>(Fz, bufA, Fr, bufD, Sbuf, nElemM);
    gemm(bufD, HID, WhBotT, KzB, Wh_b, bufB, NM, KzB);           // (r*S)@Wh_bot + bh
    gru2<<<(nElemM + 255) / 256, 256, 0, stream>>>(Fh, bufB, bufA, Sbuf, hbuf, nElemM);
  }

  // ---- phase 3: encoder attention over nodes + output head ----
  gemm(hbuf, HID, eWkT, HID, eWk_b, bufA, NM, HID);              // enc K over messages
  attn_scalar<<<rowsGrid(NM), 256, 0, stream>>>(bufA, ealpha, 32, sk, NM);
  row_mask<<<rowsGrid(NM), 256, 0, stream>>>(hbuf, flags, NM);
  gemm(hbuf, HID, eWvT, HID, eWv_b, bufB, NM, HID);              // enc V over messages
  attn_gather<<<rowsGrid(NN), 256, 0, stream>>>(agraph, sqn, sk, eabias, flags,
                                                bufB, Sbuf, NN);
  gemm(fnode, NF, WoTopT, NFp, nullptr, bufD, NN, NF);           // fnode @ Wo_top
  gemm(Sbuf, HID, WoBotT, KoBp, Wo_b, bufA, NN, KoB);            // nei_sum @ Wo_bot + b
  int nElemN = NN * HID;
  float* out0 = (float*)d_out;
  float* out1 = out0 + (size_t)nElemN;
  out_gelu<<<(nElemN + 255) / 256, 256, 0, stream>>>(bufD, bufA, maskp, out0, nElemN);
  copy_f<<<(nElemM + 255) / 256, 256, 0, stream>>>(hbuf, out1, nElemM);
}
